// TinyRnnCharModel_50869592655034
// MI455X (gfx1250) — matrix operand order repaired and checked by reading, not compiled, not run on hardware
//
#include <hip/hip_runtime.h>
#include <hip/hip_bf16.h>
#include <stdint.h>

// ---------------------------------------------------------------- constants
#define VOCAB 128
#define EMB   512
#define HID   1024
#define BB    256
#define TT    256

#define MTILES  (BB / 16)    // 16 batch-row tiles
#define NKT     (HID / 32)   // 32 K-tiles (K=32 per WMMA) per 1024
#define NNT     (HID / 16)   // 64 N-tiles over HID
#define NVT     (VOCAB / 16) // 8  N-tiles over VOCAB
#define TILE_US 512          // ushorts per packed 16x32 bf16 fragment tile
#define NTW     4            // output n-tiles per wave in step kernels

#define CHUNK_KT 8                            // k-tiles per TDM chunk
#define NCHUNK   (NKT / CHUNK_KT)             // 4 chunks per K=1024 pass
#define ROW_US   (CHUNK_KT * TILE_US)         // 4096 ushorts = 8 KB per n-row chunk
#define CHUNK_US (NTW * ROW_US)               // 16384 ushorts = 32 KB per buffer

typedef __attribute__((ext_vector_type(16))) __bf16 v16bf;
typedef __attribute__((ext_vector_type(8)))  float  v8f;
typedef __attribute__((ext_vector_type(4)))  unsigned int v4u;
typedef __attribute__((ext_vector_type(8)))  int v8i;
typedef __attribute__((ext_vector_type(4)))  int v4i;

// ---------------------------------------------------------------- helpers
__device__ __forceinline__ unsigned short f2bf(float f) {
  unsigned int u = __float_as_uint(f);
  unsigned int r = u + 0x7fffu + ((u >> 16) & 1u);   // round-to-nearest-even
  return (unsigned short)(r >> 16);
}

__device__ __forceinline__ v16bf load_frag(const unsigned short* p, int tile, int lane) {
  return *(const v16bf*)(p + (size_t)tile * TILE_US + lane * 16);
}

// A-fragment placement (CDNA5 ISA 7.12.2, 16-bit A 16x32)
__device__ __forceinline__ int a_pack_off(int m_local, int k_local) {
  int half, j;
  if (k_local < 16) { half = (k_local >= 8);  j = (k_local & 7) >> 1; }
  else              { half = (k_local >= 24); j = 4 + (((k_local - 16) & 7) >> 1); }
  int lane = m_local + half * 16;
  return lane * 16 + j * 2 + (k_local & 1);
}

__device__ __forceinline__ void store_h(unsigned short* __restrict__ h_out,
                                        int mt, int m_local, int n, float h) {
  int kt = n >> 5, kl = n & 31;
  h_out[(size_t)(mt * NKT + kt) * TILE_US + a_pack_off(m_local, kl)] = f2bf(h);
}

__device__ __forceinline__ v8f wmma_bf16(v16bf a, v16bf b, v8f c) {
  return __builtin_amdgcn_wmma_f32_16x16x32_bf16(false, a, false, b,
                                                 (short)0, c, false, false);
}

// ---------------------------------------------------------------- TDM
// Issue one Tensor-DMA: copy `rows` rows of `elems0` 4-byte elements
// (row stride `stride0` elements) from global to LDS at byte offset lds_off.
// D# layout per CDNA5 ISA 8.3/8.4.
__device__ __forceinline__ void tdm_load_rows(unsigned lds_off,
                                              const unsigned short* gsrc,
                                              unsigned elems0, unsigned rows,
                                              unsigned stride0) {
  unsigned long long ga = (unsigned long long)(uintptr_t)gsrc;
  v4u g0;
  g0.x = 1u;                                            // count=1 (valid), user mode
  g0.y = lds_off;                                       // lds_addr (bytes)
  g0.z = (unsigned)ga;                                  // global_addr[31:0]
  g0.w = (unsigned)((ga >> 32) & 0x01FFFFFFu) | (2u << 30); // addr[56:32] | type=2
  v8i g1;
  g1[0] = (int)(2u << 16);                              // wg_mask=0, data_size=4B
  g1[1] = (int)(0xFFFFu << 16);                         // tensor_dim0[15:0] (large)
  g1[2] = (int)(0x7FFFu | (0xFFFFu << 16));             // tdim0 hi | tensor_dim1 lo
  g1[3] = (int)(0x7FFFu | (elems0 << 16));              // tdim1 hi | tile_dim0
  g1[4] = (int)rows;                                    // tile_dim1 (tile_dim2=0)
  g1[5] = (int)stride0;                                 // tensor_dim0_stride[31:0]
  g1[6] = 0;                                            // stride0 hi | stride1 lo
  g1[7] = 0;
  v4i gz = {0, 0, 0, 0};
#if __has_include(<hip/amd_detail/amd_gfx1250_TDM.h>)
  v8i gz8 = {0, 0, 0, 0, 0, 0, 0, 0};
  __builtin_amdgcn_tensor_load_to_lds(g0, g1, gz, gz, gz8, 0);   // clang-23 form
#else
  __builtin_amdgcn_tensor_load_to_lds(g0, g1, gz, gz, 0);        // ROCm 7.2 form
#endif
}

// The TDM writes LDS through a descriptor the alias analysis cannot see:
// escape the LDS pointer into an opaque asm that may write memory, so the
// compiler cannot fold the subsequent ds_loads of sB to undef.
__device__ __forceinline__ void lds_written_by_tdm(const unsigned short* sB) {
  asm volatile("" :: "v"(sB) : "memory");
}

// one K=1024 GEMM pass: acc[i] += A(mt,:) x B(nt0+i,:)^T.
// B is DMA'd by the TDM into LDS in 32 KB chunks, double-buffered; wave 0
// issues the DMA and waits TENSORcnt, everyone else syncs at the barrier.
__device__ __forceinline__ void gemm_pass_tdm(const unsigned short* __restrict__ Alane,
                                              const unsigned short* __restrict__ Bbase,
                                              const unsigned short* sB,
                                              v8f* acc, int lane, int tid) {
  const unsigned stride0 = NKT * (TILE_US / 2);         // 8192 4B-elems between n-rows
  if (tid < 32) {
    tdm_load_rows(0u, Bbase, ROW_US / 2, NTW, stride0);
    __builtin_amdgcn_s_wait_tensorcnt(0);
  }
  lds_written_by_tdm(sB);
  __syncthreads();
  for (int c = 0; c < NCHUNK; ++c) {
    int buf = c & 1;
    if ((tid < 32) && (c + 1 < NCHUNK))
      tdm_load_rows((unsigned)((buf ^ 1) * (CHUNK_US * 2)),
                    Bbase + (size_t)(c + 1) * ROW_US,
                    ROW_US / 2, NTW, stride0);          // DMA overlaps the math below
    v16bf a[CHUNK_KT];
#pragma unroll
    for (int r = 0; r < CHUNK_KT; ++r)
      a[r] = *(const v16bf*)(Alane + (size_t)(c * CHUNK_KT + r) * TILE_US);
    const unsigned short* bb = sB + (size_t)buf * CHUNK_US + lane * 16;
#pragma unroll
    for (int r = 0; r < CHUNK_KT; ++r) {
#pragma unroll
      for (int i = 0; i < NTW; ++i) {
        v16bf b = *(const v16bf*)(bb + (size_t)i * ROW_US + (size_t)r * TILE_US);
        acc[i] = wmma_bf16(a[r], b, acc[i]);
      }
    }
    if (tid < 32) __builtin_amdgcn_s_wait_tensorcnt(0);
    lds_written_by_tdm(sB);
    __syncthreads();
  }
}

// ---------------------------------------------------------------- pack kernels
__global__ void pack_emb_proj(const float* __restrict__ emb,
                              const float* __restrict__ W_ih0,
                              const float* __restrict__ b_ih0,
                              float* __restrict__ emb_proj) {
  int id = blockIdx.x * blockDim.x + threadIdx.x;
  if (id >= VOCAB * HID) return;
  int v = id >> 10, n = id & (HID - 1);
  const float* e = emb + (size_t)v * EMB;
  const float* w = W_ih0 + (size_t)n * EMB;
  float s = b_ih0[n];
  for (int k = 0; k < EMB; ++k) s = fmaf(e[k], w[k], s);
  emb_proj[id] = s;
}

// Pack W[n][k] (row-major fp32) into bf16 B-fragments, tiles ordered [nt][kt].
__global__ void pack_B(const float* __restrict__ W, unsigned short* __restrict__ out,
                       int K, int KT) {
  int tile = blockIdx.x;
  int lane = threadIdx.x >> 3;
  int j    = threadIdx.x & 7;
  int nt = tile / KT, kt = tile % KT;
  int n = nt * 16 + (lane & 15);
  int k = kt * 32 + (lane >> 4) * 8 + (j >> 2) * 8 + 2 * j;
  const float* src = W + (size_t)n * K + k;
  unsigned short* dst = out + (size_t)tile * TILE_US + lane * 16 + j * 2;
  dst[0] = f2bf(src[0]);
  dst[1] = f2bf(src[1]);
}

// ---------------------------------------------------------------- RNN steps
// block = 8 waves sharing one n-tile group (TDM-staged B), covering 8 mt.
// grid: 32 blocks x 256 threads;  blockIdx = ntg*2 + mgroup
__global__ void rnn_step0(const unsigned short* __restrict__ h_prev,  // A-packed
                          const unsigned short* __restrict__ Whh,     // B-packed [NNT][NKT]
                          const float* __restrict__ emb_proj,         // [VOCAB][HID]
                          const int* __restrict__ tokens,             // [BB][TT]
                          const float* __restrict__ b_hh,             // [HID]
                          unsigned short* __restrict__ h_out,         // A-packed
                          int t) {
  __shared__ alignas(16) unsigned short sB[2 * CHUNK_US];   // 64 KB, LDS offset 0
  int tid = threadIdx.x;
  int lane = tid & 31;
  int nt0 = (blockIdx.x >> 1) * NTW;
  int mt  = (blockIdx.x & 1) * 8 + (tid >> 5);

  v8f acc[NTW] = {};
  gemm_pass_tdm(h_prev + (size_t)mt * NKT * TILE_US + lane * 16,
                Whh + (size_t)nt0 * NKT * TILE_US, sB, acc, lane, tid);

  int half = lane >> 4, nloc = lane & 15;
  int toks[8];
#pragma unroll
  for (int j = 0; j < 8; ++j)
    toks[j] = tokens[(mt * 16 + j + half * 8) * TT + t];

#pragma unroll
  for (int i = 0; i < NTW; ++i) {
    int n = (nt0 + i) * 16 + nloc;
    float bias = b_hh[n];
#pragma unroll
    for (int j = 0; j < 8; ++j) {
      int m_local = j + half * 8;
      float v = acc[i][j] + emb_proj[(size_t)toks[j] * HID + n] + bias;
      store_h(h_out, mt, m_local, n, tanhf(v));
    }
  }
}

// layer 1 (fused input projection, K = 2048 via two passes)
__global__ void rnn_step1(const unsigned short* __restrict__ h0_cur,   // A-packed
                          const unsigned short* __restrict__ h1_prev,  // A-packed
                          const unsigned short* __restrict__ Wih,      // B-packed
                          const unsigned short* __restrict__ Whh,      // B-packed
                          const float* __restrict__ b_ih,
                          const float* __restrict__ b_hh,
                          unsigned short* __restrict__ h_out) {
  __shared__ alignas(16) unsigned short sB[2 * CHUNK_US];
  int tid = threadIdx.x;
  int lane = tid & 31;
  int nt0 = (blockIdx.x >> 1) * NTW;
  int mt  = (blockIdx.x & 1) * 8 + (tid >> 5);

  v8f acc[NTW] = {};
  gemm_pass_tdm(h0_cur  + (size_t)mt * NKT * TILE_US + lane * 16,
                Wih + (size_t)nt0 * NKT * TILE_US, sB, acc, lane, tid);
  gemm_pass_tdm(h1_prev + (size_t)mt * NKT * TILE_US + lane * 16,
                Whh + (size_t)nt0 * NKT * TILE_US, sB, acc, lane, tid);

  int half = lane >> 4, nloc = lane & 15;
#pragma unroll
  for (int i = 0; i < NTW; ++i) {
    int n = (nt0 + i) * 16 + nloc;
    float bias = b_ih[n] + b_hh[n];
#pragma unroll
    for (int j = 0; j < 8; ++j) {
      int m_local = j + half * 8;
      store_h(h_out, mt, m_local, n, tanhf(acc[i][j] + bias));
    }
  }
}

// out[b][v] = h1_last[b] . W_out[v] + b_out[v]    (128 waves, 1 tile each)
__global__ void out_head(const unsigned short* __restrict__ h1,    // A-packed
                         const unsigned short* __restrict__ Wout,  // B-packed [NVT][NKT]
                         const float* __restrict__ b_out,
                         float* __restrict__ out) {
  int lane = threadIdx.x & 31;
  int wave = blockIdx.x * (blockDim.x >> 5) + (threadIdx.x >> 5);
  int mt = wave >> 3;
  int nt = wave & 7;

  v8f acc = {};
  for (int kt = 0; kt < NKT; ++kt) {
    v16bf a = load_frag(h1, mt * NKT + kt, lane);
    v16bf b = load_frag(Wout, nt * NKT + kt, lane);
    acc = wmma_bf16(a, b, acc);
  }
  int half = lane >> 4;
  int n = nt * 16 + (lane & 15);
  float bias = b_out[n];
#pragma unroll
  for (int j = 0; j < 8; ++j) {
    int brow = mt * 16 + j + half * 8;
    out[(size_t)brow * VOCAB + n] = acc[j] + bias;
  }
}

// ---------------------------------------------------------------- launch
extern "C" void kernel_launch(void* const* d_in, const int* in_sizes, int n_in,
                              void* d_out, int out_size, void* d_ws, size_t ws_size,
                              hipStream_t stream) {
  const int*   tokens = (const int*)  d_in[0];
  const float* emb    = (const float*)d_in[1];
  const float* W_ih0  = (const float*)d_in[2];
  const float* W_hh0  = (const float*)d_in[3];
  const float* b_ih0  = (const float*)d_in[4];
  const float* b_hh0  = (const float*)d_in[5];
  const float* W_ih1  = (const float*)d_in[6];
  const float* W_hh1  = (const float*)d_in[7];
  const float* b_ih1  = (const float*)d_in[8];
  const float* b_hh1  = (const float*)d_in[9];
  const float* W_out  = (const float*)d_in[10];
  const float* b_out  = (const float*)d_in[11];
  float* out = (float*)d_out;

  char* ws = (char*)d_ws;
  size_t off = 0;
  auto alloc = [&](size_t bytes) -> void* {
    void* p = ws + off;
    off = (off + bytes + 255) & ~(size_t)255;
    return p;
  };
  const size_t WTILE_BYTES = (size_t)NNT * NKT * TILE_US * 2;     // 2 MB
  const size_t HBYTES      = (size_t)MTILES * NKT * TILE_US * 2;  // 512 KB

  unsigned short* Whh0_p = (unsigned short*)alloc(WTILE_BYTES);
  unsigned short* Wih1_p = (unsigned short*)alloc(WTILE_BYTES);
  unsigned short* Whh1_p = (unsigned short*)alloc(WTILE_BYTES);
  unsigned short* Wout_p = (unsigned short*)alloc((size_t)NVT * NKT * TILE_US * 2);
  float*          embP   = (float*)alloc((size_t)VOCAB * HID * 4);
  unsigned short* h0[2]  = { (unsigned short*)alloc(HBYTES), (unsigned short*)alloc(HBYTES) };
  unsigned short* h1[2]  = { (unsigned short*)alloc(HBYTES), (unsigned short*)alloc(HBYTES) };

  // per-call precompute: embedding projection + bf16 weight packing
  pack_emb_proj<<<(VOCAB * HID + 255) / 256, 256, 0, stream>>>(emb, W_ih0, b_ih0, embP);
  pack_B<<<NNT * NKT, 256, 0, stream>>>(W_hh0, Whh0_p, HID, NKT);
  pack_B<<<NNT * NKT, 256, 0, stream>>>(W_ih1, Wih1_p, HID, NKT);
  pack_B<<<NNT * NKT, 256, 0, stream>>>(W_hh1, Whh1_p, HID, NKT);
  pack_B<<<NVT * NKT, 256, 0, stream>>>(W_out, Wout_p, HID, NKT);

  hipMemsetAsync(h0[0], 0, HBYTES, stream);
  hipMemsetAsync(h1[0], 0, HBYTES, stream);

  // sequential scan: stream order provides the grid-wide dependency per step
  int cur = 0;
  for (int t = 0; t < TT; ++t) {
    int nxt = cur ^ 1;
    rnn_step0<<<32, 256, 0, stream>>>(h0[cur], Whh0_p, embP, tokens, b_hh0, h0[nxt], t);
    rnn_step1<<<32, 256, 0, stream>>>(h0[nxt], h1[cur], Wih1_p, Whh1_p, b_ih1, b_hh1, h1[nxt]);
    cur = nxt;
  }
  out_head<<<16, 256, 0, stream>>>(h1[cur], Wout_p, b_out, out);
}
